// AttentionModule_83863531422241
// MI455X (gfx1250) — matrix operand order repaired and checked by reading, not compiled, not run on hardware
//
#include <hip/hip_runtime.h>
#include <hip/hip_bf16.h>
#include <stdint.h>

#define B_    8
#define T_    1024
#define DIM_  512
#define H_    8
#define HD_   64
#define P_    2047

typedef __attribute__((ext_vector_type(16))) __bf16 v16bf;
typedef __attribute__((ext_vector_type(8)))  float  v8f;
typedef __attribute__((ext_vector_type(4))) unsigned v4u;
typedef __attribute__((ext_vector_type(8)))  int     v8i;
typedef __attribute__((ext_vector_type(4)))  int     v4i;

#if __has_builtin(__builtin_amdgcn_tensor_load_to_lds)
#define HAVE_TDM 1
#else
#define HAVE_TDM 0
#endif

union Frag { v16bf v; unsigned u[8]; __bf16 h[16]; };

__device__ __forceinline__ __bf16 f2bf(float f) {
  unsigned u = __builtin_bit_cast(unsigned, f);
  unsigned r = (u + 0x7FFFu + ((u >> 16) & 1u)) >> 16;
  unsigned short s = (unsigned short)r;
  return __builtin_bit_cast(__bf16, s);
}

__device__ __forceinline__ v8f wmma_bf16(v16bf a, v16bf b, v8f c) {
  return __builtin_amdgcn_wmma_f32_16x16x32_bf16(false, a, false, b, (short)0, c,
                                                 false, false);
}

// ---- TDM: DMA a 2D tile (rows x rowBytes, global stride strideBytes) into LDS.
// D# built per CDNA5 ISA ch.8: data_size=8B, 2D tile, groups 2/3 zero.
// tensorRows < rows zero-fills the OOB tail rows.
__device__ __forceinline__ void tdm_load_2d(unsigned lds_off, const void* gsrc,
                                            unsigned rowBytes, unsigned rows,
                                            unsigned strideBytes, unsigned tensorRows) {
#if HAVE_TDM
  uint64_t ga = (uint64_t)gsrc;
  unsigned u0 = rowBytes >> 3;      // dim0 length/tile in 8B units
  unsigned su = strideBytes >> 3;   // dim0 stride in 8B units
  v4u g0; v8i g1; v4i gz4; v8i gz8;
  gz4[0] = 0; gz4[1] = 0; gz4[2] = 0; gz4[3] = 0;
  gz8[0] = 0; gz8[1] = 0; gz8[2] = 0; gz8[3] = 0;
  gz8[4] = 0; gz8[5] = 0; gz8[6] = 0; gz8[7] = 0;
  g0[0] = 1u;                                    // count=1, user, no gather
  g0[1] = lds_off;                               // lds_addr
  g0[2] = (unsigned)ga;                          // global_addr[31:0]
  g0[3] = (unsigned)((ga >> 32) & 0x01FFFFFFu) | (2u << 30);  // addr[56:32] | type=2
  g1[0] = (int)(3u << 16);                       // wg_mask=0, data_size=8B
  g1[1] = (int)(u0 << 16);                       // tensor_dim0[15:0]
  g1[2] = (int)((u0 >> 16) | ((tensorRows & 0xFFFFu) << 16)); // dim0 hi | dim1 lo
  g1[3] = (int)((tensorRows >> 16) | (u0 << 16));             // dim1 hi | tile_dim0
  g1[4] = (int)rows;                             // tile_dim1, tile_dim2=0
  g1[5] = (int)su;                               // tensor_dim0_stride[31:0]
  g1[6] = 0;
  g1[7] = 0;
  __builtin_amdgcn_tensor_load_to_lds(g0, g1, gz4, gz4, gz8, 0);
#endif
}

__device__ __forceinline__ void tdm_wait() {
#if HAVE_TDM
  __builtin_amdgcn_s_wait_tensorcnt(0);
#endif
}

__device__ __forceinline__ unsigned lds_offset_of(const void* p) {
  return (unsigned)(uintptr_t)p;  // flat LDS address: low 32 bits are the LDS offset
}

// A-fragment 16x32 bf16 (M x K). Row passed per-lane; K pairs contiguous.
__device__ __forceinline__ v16bf load_a_frag(const __bf16* A, int row, int ld, int k0) {
  Frag f;
  int lane = threadIdx.x & 31;
  int half = lane >> 4;
  const __bf16* base = A + (size_t)row * ld + k0 + half * 16;
  #pragma unroll
  for (int v = 0; v < 8; ++v) {
    int k = v * 2;
    f.u[v] = *(const unsigned*)(base + k);
  }
  return f.v;
}

// B-fragment 32x16 bf16 (K x N) from memory laid out [n][k].
__device__ __forceinline__ v16bf load_b_nmajor(const __bf16* Bp, int n, int ld, int k0) {
  Frag f;
  int lane = threadIdx.x & 31;
  int half = lane >> 4;
  const __bf16* base = Bp + (size_t)n * ld + k0 + half * 16;
  #pragma unroll
  for (int v = 0; v < 8; ++v) f.u[v] = *(const unsigned*)(base + 2 * v);
  return f.v;
}

// B-fragment 32x16 bf16 from memory laid out [k][n] (k strided by ld).
__device__ __forceinline__ v16bf load_b_kmajor(const __bf16* Bp, int ld, int n) {
  Frag f;
  int lane = threadIdx.x & 31;
  int half = lane >> 4;
  const __bf16* base = Bp + (size_t)(half * 16) * ld + n;
  #pragma unroll
  for (int v = 0; v < 8; ++v) {
    f.h[2 * v]     = base[(size_t)(2 * v) * ld];
    f.h[2 * v + 1] = base[(size_t)(2 * v + 1) * ld];
  }
  return f.v;
}

// ---------------- prep kernels ----------------
__global__ __launch_bounds__(256) void prep_scale_bias(const float* __restrict__ x,
                                                       const float* __restrict__ sc,
                                                       const float* __restrict__ bi,
                                                       __bf16* __restrict__ out) {
  size_t i = (size_t)blockIdx.x * 256 + threadIdx.x;
  int d = (int)(i & (DIM_ - 1));
  out[i] = f2bf(x[i] * sc[d] + bi[d]);
}

__global__ __launch_bounds__(256) void prep_tobf(const float* __restrict__ src,
                                                 __bf16* __restrict__ dst) {
  size_t i = (size_t)blockIdx.x * 256 + threadIdx.x;
  dst[i] = f2bf(src[i]);
}

__global__ __launch_bounds__(256) void prep_wT(const float* __restrict__ W,
                                               __bf16* __restrict__ Wt) {
  size_t i = (size_t)blockIdx.x * 256 + threadIdx.x;
  int k = (int)(i >> 9), n = (int)(i & (DIM_ - 1));
  Wt[(size_t)n * DIM_ + k] = f2bf(W[i]);
}

// ---------------- bf16 WMMA GEMM: C[M,512] = A[M,512] @ W ----------------
// Block: 32 rows of A staged to LDS by TDM; 8 waves cover N=512 (64 cols each).
__global__ __launch_bounds__(256) void gemm512(const __bf16* __restrict__ A,
                                               const __bf16* __restrict__ Wt,
                                               const float* __restrict__ bias,
                                               const float* __restrict__ extraU,
                                               const float* __restrict__ extraV,
                                               const float* __restrict__ residual,
                                               __bf16* __restrict__ outU,
                                               __bf16* __restrict__ outV,
                                               float* __restrict__ outF,
                                               int M) {
  __shared__ __align__(16) __bf16 Atile[32][DIM_];   // 32 KB

  int wave = threadIdx.x >> 5;
  int lane = threadIdx.x & 31;
  int l16 = lane & 15, half = lane >> 4;
  int m0 = blockIdx.x * 32;
  int n0 = wave * 64;

#if HAVE_TDM
  if (threadIdx.x == 0) {
    unsigned remain = (unsigned)(M - m0);
    tdm_load_2d(lds_offset_of(&Atile[0][0]), A + (size_t)m0 * DIM_,
                DIM_ * 2, 32, DIM_ * 2, remain);
    tdm_wait();
  }
  __syncthreads();
#else
  for (int i = threadIdx.x; i < 32 * (DIM_ / 8); i += 256) {
    int r = i / (DIM_ / 8), c = (i % (DIM_ / 8)) * 8;
    int rg = m0 + r < M ? m0 + r : M - 1;
    *(ulong2*)&Atile[r][c] = *(const ulong2*)(A + (size_t)rg * DIM_ + c);
  }
  __syncthreads();
#endif

  v8f acc[2][4] = {};
  for (int k0 = 0; k0 < DIM_; k0 += 32) {
    v16bf a0 = load_a_frag(&Atile[0][0], l16, DIM_, k0);
    v16bf a1 = load_a_frag(&Atile[0][0], 16 + l16, DIM_, k0);
    #pragma unroll
    for (int j = 0; j < 4; ++j) {
      v16bf b = load_b_nmajor(Wt, n0 + j * 16 + l16, DIM_, k0);
      acc[0][j] = wmma_bf16(a0, b, acc[0][j]);
      acc[1][j] = wmma_bf16(a1, b, acc[1][j]);
    }
  }

  #pragma unroll
  for (int rt = 0; rt < 2; ++rt) {
    #pragma unroll
    for (int j = 0; j < 4; ++j) {
      #pragma unroll
      for (int r = 0; r < 8; ++r) {
        int rr = m0 + rt * 16 + r + 8 * half;
        if (rr >= M) continue;
        int cc = n0 + j * 16 + l16;
        float val = acc[rt][j][r] + (bias ? bias[cc] : 0.0f);
        size_t o = (size_t)rr * DIM_ + cc;
        if (outU) outU[o] = f2bf(val + (extraU ? extraU[cc] : 0.0f));
        if (outV) outV[o] = f2bf(val + extraV[cc]);
        if (outF) outF[o] = val + (residual ? residual[o] : 0.0f);
      }
    }
  }
}

// ---------------- fused flash attention with relative-shift band ----------------
// 8 waves / block, each wave owns 16 query rows; all waves share one (b,h).
// Per 32-key chunk the block TDM-stages (double-buffered, overlapped with compute):
//   K chunk 32x64, V chunk 32x64, and the union positional band P chunk 160x64
//   (the 8 per-wave 47-row bands are offset by 16 rows each -> one 159-row window).
__global__ __launch_bounds__(256) void attn_kernel(const __bf16* __restrict__ qu,
                                                   const __bf16* __restrict__ qv,
                                                   const __bf16* __restrict__ kb,
                                                   const __bf16* __restrict__ vb,
                                                   const __bf16* __restrict__ pb,
                                                   const unsigned char* __restrict__ mask,
                                                   __bf16* __restrict__ ctx) {
  __shared__ __align__(16) float  bandBuf[8][16][48];     // 24 KB
  __shared__ __align__(16) __bf16 pBuf[8][16][32];        //  8 KB
  __shared__ __align__(16) __bf16 Ktile[2][32][HD_];      //  8 KB
  __shared__ __align__(16) __bf16 Vtile[2][32][HD_];      //  8 KB
  __shared__ __align__(16) __bf16 Ptile[2][160][HD_];     // 40 KB

  int wave = threadIdx.x >> 5;
  int lane = threadIdx.x & 31;
  int l16 = lane & 15, half = lane >> 4;

  int id = blockIdx.x * 8 + wave;
  int b  = id / (H_ * (T_ / 16));
  int h  = (id / (T_ / 16)) % H_;
  int t0 = (id % (T_ / 16)) * 16;
  int t0blk = ((blockIdx.x * 8) % (T_ / 16)) * 16;   // t0 of wave 0 (uniform)

  const __bf16* quB = qu + (size_t)b * T_ * DIM_ + h * HD_;
  const __bf16* qvB = qv + (size_t)b * T_ * DIM_ + h * HD_;
  const __bf16* kB  = kb + (size_t)b * T_ * DIM_ + h * HD_;
  const __bf16* vB  = vb + (size_t)b * T_ * DIM_ + h * HD_;
  const __bf16* pB  = pb + (size_t)b * P_ * DIM_ + h * HD_;
  const unsigned char* mB = mask + (size_t)b * T_ * T_ + (size_t)t0 * T_;

#if HAVE_TDM
  auto stage = [&](int ci) {
    int ss0 = ci * 32, buf = ci & 1;
    tdm_load_2d(lds_offset_of(&Ktile[buf][0][0]), kB + (size_t)ss0 * DIM_,
                HD_ * 2, 32, DIM_ * 2, 32);
    tdm_load_2d(lds_offset_of(&Vtile[buf][0][0]), vB + (size_t)ss0 * DIM_,
                HD_ * 2, 32, DIM_ * 2, 32);
    int pbase = ss0 + (T_ - 128) - t0blk;            // union band start, >= 0
    tdm_load_2d(lds_offset_of(&Ptile[buf][0][0]), pB + (size_t)pbase * DIM_,
                HD_ * 2, 160, DIM_ * 2, (unsigned)(P_ - pbase));
  };
#endif

  v16bf quF[2], qvF[2];
  #pragma unroll
  for (int kc = 0; kc < 2; ++kc) {
    quF[kc] = load_a_frag(quB, t0 + l16, DIM_, kc * 32);
    qvF[kc] = load_a_frag(qvB, t0 + l16, DIM_, kc * 32);
  }

  float mrow[8], lrow[8];
  v8f ctxacc[4] = {};
  #pragma unroll
  for (int r = 0; r < 8; ++r) { mrow[r] = -3.0e38f; lrow[r] = 0.0f; }

#if HAVE_TDM
  if (threadIdx.x == 0) stage(0);
#endif

  const int NC = T_ / 32;
  for (int ci = 0; ci < NC; ++ci) {
    int ss0 = ci * 32, buf = ci & 1;
#if HAVE_TDM
    if (threadIdx.x == 0) tdm_wait();   // chunk ci resident in LDS
    __syncthreads();                    // publish buf to all waves
    if (threadIdx.x == 0 && ci + 1 < NC) stage(ci + 1);  // overlap DMA w/ compute
#else
    __syncthreads();
    {
      int pbase = ss0 + (T_ - 128) - t0blk;
      for (int i = threadIdx.x; i < 32 * (HD_ / 8); i += 256) {
        int r = i / (HD_ / 8), c = (i % (HD_ / 8)) * 8;
        *(ulong2*)&Ktile[buf][r][c] = *(const ulong2*)(kB + (size_t)(ss0 + r) * DIM_ + c);
        *(ulong2*)&Vtile[buf][r][c] = *(const ulong2*)(vB + (size_t)(ss0 + r) * DIM_ + c);
      }
      for (int i = threadIdx.x; i < 160 * (HD_ / 8); i += 256) {
        int r = i / (HD_ / 8), c = (i % (HD_ / 8)) * 8;
        int pr = pbase + r; pr = pr > P_ - 1 ? P_ - 1 : pr;
        *(ulong2*)&Ptile[buf][r][c] = *(const ulong2*)(pB + (size_t)pr * DIM_ + c);
      }
    }
    __syncthreads();
#endif

    // ---- content scores from LDS K: two 16x16 tiles, K = 64 ----
    v8f S0 = {}, S1 = {};
    #pragma unroll
    for (int kc = 0; kc < 2; ++kc) {
      S0 = wmma_bf16(quF[kc], load_b_nmajor(&Ktile[buf][0][0], l16, HD_, kc * 32), S0);
      S1 = wmma_bf16(quF[kc], load_b_nmajor(&Ktile[buf][0][0], 16 + l16, HD_, kc * 32), S1);
    }

    // ---- positional band from LDS P: this wave's rows start at 112 - 16*wave ----
    #pragma unroll
    for (int tile = 0; tile < 3; ++tile) {
      v8f Pp = {};
      int jl = 112 - 16 * wave + tile * 16 + l16;
      #pragma unroll
      for (int kc = 0; kc < 2; ++kc)
        Pp = wmma_bf16(qvF[kc], load_b_nmajor(&Ptile[buf][0][0], jl, HD_, kc * 32), Pp);
      #pragma unroll
      for (int r = 0; r < 8; ++r)
        bandBuf[wave][r + 8 * half][tile * 16 + l16] = Pp[r];
    }

    // ---- assemble scores, streaming softmax ----
    float corr[8];
    #pragma unroll
    for (int r = 0; r < 8; ++r) {
      int rloc = r + 8 * half;
      float s0 = (S0[r] + bandBuf[wave][rloc][l16 + 15 - rloc]) * 0.125f;
      float s1 = (S1[r] + bandBuf[wave][rloc][l16 + 31 - rloc]) * 0.125f;
      if (!mB[(size_t)rloc * T_ + ss0 + l16])      s0 = -3.0e38f;
      if (!mB[(size_t)rloc * T_ + ss0 + 16 + l16]) s1 = -3.0e38f;
      float mx = fmaxf(s0, s1);
      #pragma unroll
      for (int o = 1; o < 16; o <<= 1) mx = fmaxf(mx, __shfl_xor(mx, o, 32));
      float mnew = fmaxf(mrow[r], mx);
      float cf = __expf(mrow[r] - mnew);
      float p0 = __expf(s0 - mnew), p1 = __expf(s1 - mnew);
      float sum = p0 + p1;
      #pragma unroll
      for (int o = 1; o < 16; o <<= 1) sum += __shfl_xor(sum, o, 32);
      lrow[r] = lrow[r] * cf + sum;
      mrow[r] = mnew;
      corr[r] = cf;
      pBuf[wave][rloc][l16]      = f2bf(p0);
      pBuf[wave][rloc][16 + l16] = f2bf(p1);
    }
    #pragma unroll
    for (int nt = 0; nt < 4; ++nt)
      #pragma unroll
      for (int r = 0; r < 8; ++r) ctxacc[nt][r] *= corr[r];

    // ---- ctx += P(16x32) @ V(32x64), V from LDS ----
    Frag pf;
    {
      const __bf16* pr = &pBuf[wave][l16][half * 16];
      #pragma unroll
      for (int v = 0; v < 8; ++v) {
        int k = v * 2;
        pf.u[v] = *(const unsigned*)(pr + k);
      }
    }
    #pragma unroll
    for (int nt = 0; nt < 4; ++nt)
      ctxacc[nt] = wmma_bf16(pf.v, load_b_kmajor(&Vtile[buf][0][0], HD_, nt * 16 + l16),
                             ctxacc[nt]);

    __syncthreads();   // all reads of buf done before TDM of ci+2 overwrites it
  }

  // ---- normalize + store ctx ----
  #pragma unroll
  for (int r = 0; r < 8; ++r) {
    int rloc = r + 8 * half;
    float inv = 1.0f / fmaxf(lrow[r], 1e-30f);
    #pragma unroll
    for (int nt = 0; nt < 4; ++nt)
      ctx[((size_t)b * T_ + t0 + rloc) * DIM_ + h * HD_ + nt * 16 + l16] =
          f2bf(ctxacc[nt][r] * inv);
  }
}

// ---------------- host launch ----------------
extern "C" void kernel_launch(void* const* d_in, const int* in_sizes, int n_in,
                              void* d_out, int out_size, void* d_ws, size_t ws_size,
                              hipStream_t stream) {
  const float* x    = (const float*)d_in[0];
  const float* pos  = (const float*)d_in[1];
  const unsigned char* mask = (const unsigned char*)d_in[2];
  const float* scale = (const float*)d_in[3];
  const float* bias  = (const float*)d_in[4];
  const float* Wq = (const float*)d_in[5];
  const float* bq = (const float*)d_in[6];
  const float* Wk = (const float*)d_in[7];
  const float* bk = (const float*)d_in[8];
  const float* Wv = (const float*)d_in[9];
  const float* bv = (const float*)d_in[10];
  const float* Wpos = (const float*)d_in[11];
  const float* Wout = (const float*)d_in[12];
  const float* bout = (const float*)d_in[13];
  const float* pbu  = (const float*)d_in[14];
  const float* pbv  = (const float*)d_in[15];
  float* out = (float*)d_out;

  const size_t BT = (size_t)B_ * T_, BP = (size_t)B_ * P_;
  size_t off = 0;
  auto alloc = [&](size_t bytes) -> void* {
    void* p = (char*)d_ws + off;
    off += (bytes + 255) & ~(size_t)255;
    return p;
  };
  __bf16* xt    = (__bf16*)alloc(BT * DIM_ * 2);
  __bf16* posb  = (__bf16*)alloc(BP * DIM_ * 2);
  __bf16* WqT   = (__bf16*)alloc((size_t)DIM_ * DIM_ * 2);
  __bf16* WkT   = (__bf16*)alloc((size_t)DIM_ * DIM_ * 2);
  __bf16* WvT   = (__bf16*)alloc((size_t)DIM_ * DIM_ * 2);
  __bf16* WposT = (__bf16*)alloc((size_t)DIM_ * DIM_ * 2);
  __bf16* WoutT = (__bf16*)alloc((size_t)DIM_ * DIM_ * 2);
  __bf16* q_u   = (__bf16*)alloc(BT * DIM_ * 2);
  __bf16* q_v   = (__bf16*)alloc(BT * DIM_ * 2);
  __bf16* k_b   = (__bf16*)alloc(BT * DIM_ * 2);
  __bf16* v_b   = (__bf16*)alloc(BT * DIM_ * 2);
  __bf16* p_b   = (__bf16*)alloc(BP * DIM_ * 2);
  __bf16* ctx_b = (__bf16*)alloc(BT * DIM_ * 2);
  (void)ws_size; (void)in_sizes; (void)n_in; (void)out_size;

  prep_scale_bias<<<(unsigned)(BT * DIM_ / 256), 256, 0, stream>>>(x, scale, bias, xt);
  prep_tobf<<<(unsigned)(BP * DIM_ / 256), 256, 0, stream>>>(pos, posb);
  prep_wT<<<DIM_ * DIM_ / 256, 256, 0, stream>>>(Wq, WqT);
  prep_wT<<<DIM_ * DIM_ / 256, 256, 0, stream>>>(Wk, WkT);
  prep_wT<<<DIM_ * DIM_ / 256, 256, 0, stream>>>(Wv, WvT);
  prep_wT<<<DIM_ * DIM_ / 256, 256, 0, stream>>>(Wpos, WposT);
  prep_wT<<<DIM_ * DIM_ / 256, 256, 0, stream>>>(Wout, WoutT);

  int Mx = (int)BT, Mp = (int)BP;
  gemm512<<<(Mx + 31) / 32, 256, 0, stream>>>(xt, WqT, bq, pbu, pbv, nullptr,
                                              q_u, q_v, nullptr, Mx);
  gemm512<<<(Mx + 31) / 32, 256, 0, stream>>>(xt, WkT, bk, nullptr, nullptr, nullptr,
                                              k_b, nullptr, nullptr, Mx);
  gemm512<<<(Mx + 31) / 32, 256, 0, stream>>>(xt, WvT, bv, nullptr, nullptr, nullptr,
                                              v_b, nullptr, nullptr, Mx);
  gemm512<<<(Mp + 31) / 32, 256, 0, stream>>>(posb, WposT, nullptr, nullptr, nullptr,
                                              nullptr, p_b, nullptr, nullptr, Mp);

  attn_kernel<<<(B_ * H_ * (T_ / 16)) / 8, 256, 0, stream>>>(q_u, q_v, k_b, v_b, p_b,
                                                             mask, ctx_b);

  gemm512<<<(Mx + 31) / 32, 256, 0, stream>>>(ctx_b, WoutT, bout, nullptr, nullptr,
                                              x, nullptr, nullptr, out, Mx);
}
